// GNN_70660801954146
// MI455X (gfx1250) — matrix operand order repaired and checked by reading, not compiled, not run on hardware
//
#include <hip/hip_runtime.h>
#include <cstdint>
#include <cstddef>

typedef __bf16 bf16_t;
typedef __attribute__((ext_vector_type(8)))  __bf16 v8bf;
typedef __attribute__((ext_vector_type(16))) __bf16 v16bf;
typedef __attribute__((ext_vector_type(8)))  float  v8f;
typedef unsigned int v4u  __attribute__((ext_vector_type(4)));
typedef int          v8i_t __attribute__((ext_vector_type(8)));
typedef int          v4i_t __attribute__((ext_vector_type(4)));

#if defined(__has_builtin)
#  if __has_builtin(__builtin_amdgcn_tensor_load_to_lds) && \
      __has_builtin(__builtin_amdgcn_s_wait_tensorcnt)
#    define USE_TDM 1
#  else
#    define USE_TDM 0
#  endif
#else
#  define USE_TDM 0
#endif

// ---------------------------------------------------------------------------
// CDNA5 packed-bf16 global atomic add (non-returning, device scope).
// Halves atomic traffic vs fp32 adds for the edge-scatter phase.
// ---------------------------------------------------------------------------
__device__ __forceinline__ void atomic_pk_add_bf16(bf16_t* addr, uint32_t pk) {
  asm volatile("global_atomic_pk_add_bf16 %0, %1, off scope:SCOPE_DEV"
               :: "v"((uint64_t)(uintptr_t)addr), "v"(pk)
               : "memory");
}

// ---------------------------------------------------------------------------
// Convert fp32 input features into the h-region (cols 0:128) of A0 [N,640].
// ---------------------------------------------------------------------------
__global__ void __launch_bounds__(256) cvt_x_kernel(const float* __restrict__ x,
                                                    bf16_t* __restrict__ A, int total) {
  int idx = blockIdx.x * 256 + threadIdx.x;
  if (idx >= total) return;
  int n = idx >> 7;          // /128
  int i = idx & 127;
  A[(long)n * 640 + i] = (bf16_t)x[idx];
}

// ---------------------------------------------------------------------------
// Zero the aggregation region (cols DIN:5*DIN) of A, 16 bytes per thread.
// ---------------------------------------------------------------------------
template <int DIN>
__global__ void __launch_bounds__(256) zero_agg_kernel(bf16_t* __restrict__ A, int total) {
  const int HP = DIN / 2;    // uint4 chunks per row (4*DIN bf16 = 8*DIN bytes)
  int idx = blockIdx.x * 256 + threadIdx.x;
  if (idx >= total) return;
  int row = idx / HP;
  int t   = idx % HP;
  uint4* p = (uint4*)(A + (long)row * (5 * DIN) + DIN) + t;
  *p = uint4{0u, 0u, 0u, 0u};
}

// ---------------------------------------------------------------------------
// Pack [lin_w | conv_w[0..3]] into bf16 Wcat[dout, 5*DIN], row-major.
// ---------------------------------------------------------------------------
template <int DIN, int DOUT>
__global__ void __launch_bounds__(256) pack_w_kernel(const float* __restrict__ lin,
                                                     const float* __restrict__ conv,
                                                     bf16_t* __restrict__ W) {
  const int K = 5 * DIN;
  int idx = blockIdx.x * 256 + threadIdx.x;
  if (idx >= DOUT * K) return;
  int o = idx / K;
  int i = idx % K;
  int region = i / DIN;      // 0 = linear, 1..4 = colour c = region-1
  float v = (region == 0)
              ? lin[(long)o * DIN + i]
              : conv[((long)(region - 1) * DOUT + o) * DIN + (i - region * DIN)];
  W[idx] = (bf16_t)v;
}

// ---------------------------------------------------------------------------
// Edge scatter: gather bf16 pair from A[src] h-region, pk-atomic-add into
// A[dst] colour bucket. One thread per (edge, pair).
// ---------------------------------------------------------------------------
template <int DIN>
__global__ void __launch_bounds__(256) scatter_kernel(const int* __restrict__ src,
                                                      const int* __restrict__ dst,
                                                      const int* __restrict__ col,
                                                      bf16_t* __restrict__ A, int E) {
  const int HP = DIN / 2;
  int idx = blockIdx.x * 256 + threadIdx.x;
  int e = idx / HP;
  int p = idx % HP;
  if (e >= E) return;
  int s = src[e];
  int d = dst[e];
  int c = col[e];
  uint32_t v = *(const uint32_t*)(A + (long)s * (5 * DIN) + 2 * p);
  bf16_t* t = A + (long)d * (5 * DIN) + (1 + c) * DIN + 2 * p;
  atomic_pk_add_bf16(t, v);
}

// ---------------------------------------------------------------------------
// Fused GEMM: out[N,DOUT] = act( A[N,5*DIN] @ Wcat[DOUT,5*DIN]^T + bias ).
//
// The 16-row A stripe is staged into LDS once per block by the Tensor Data
// Mover (TDM descriptor with LDS padding: pad 4 dwords every 64 dwords ->
// phys(d) = d + 4*(d>>6); lanes' b128 reads start at banks 20*r mod 64 (40*r for DIN=256), all
// distinct only for DIN=128 -> conflict-free there, and offsets stay 16B aligned). Waves then build
// WMMA A-fragments via ds_load_b128 and B-fragments via 32B global loads.
// ---------------------------------------------------------------------------
template <int DIN, int DOUT, bool FINAL>
__global__ void __launch_bounds__(256) gemm_kernel(const bf16_t* __restrict__ A,
                                                   const bf16_t* __restrict__ W,
                                                   const float* __restrict__ bias,
                                                   void* __restrict__ outp) {
  const int K        = 5 * DIN;
  const int ROWDW    = K / 2;                       // dwords per logical row (320/640)
  const int PADROWDW = ROWDW + (ROWDW / 64) * 4;    // padded row stride in dwords
  const int NT       = DOUT / 128;                  // n-tiles per wave

  __shared__ uint32_t Atile[16 * PADROWDW];

  int tid  = threadIdx.x;
  int wave = tid >> 5;
  int lane = tid & 31;
  int r = lane & 15;
  int g = lane >> 4;
  long m0 = (long)blockIdx.x * 16;

#if USE_TDM
  if (tid < 32) {
    uint64_t ga  = (uint64_t)(uintptr_t)(A + m0 * K);
    uint32_t lds = (uint32_t)(uintptr_t)(&Atile[0]);
    v4u g0;
    g0.x = 1u;                                        // count=1, user descriptor
    g0.y = lds;                                       // lds_addr [63:32]
    g0.z = (uint32_t)ga;                              // global_addr [95:64]
    g0.w = ((uint32_t)(ga >> 32) & 0x01FFFFFFu)       // global_addr [120:96]
           | 0x80000000u;                             // type=2 ("image")
    v8i_t g1;
    g1[0] = (int)((2u << 16)        // data_size = 4B
                | (1u << 20)        // pad_enable
                | (5u << 22)        // pad_interval: every 64 dwords
                | (3u << 25));      // pad_amount: 4 dwords
    g1[1] = (int)(((uint32_t)ROWDW & 0xFFFFu) << 16); // tensor_dim0 [15:0]
    g1[2] = (int)(((uint32_t)ROWDW >> 16) | (16u << 16)); // dim0 hi | tensor_dim1=16
    g1[3] = (int)(((uint32_t)ROWDW) << 16);           // tile_dim0 = ROWDW
    g1[4] = 16;                                       // tile_dim1=16, tile_dim2=0
    g1[5] = ROWDW;                                    // tensor_dim0_stride lo
    g1[6] = 0;                                        // stride hi / dim1_stride lo
    g1[7] = 0;
    v4i_t gz = {0, 0, 0, 0};
#if defined(__clang_major__) && (__clang_major__ >= 23)
    v8i_t gz8 = {0, 0, 0, 0, 0, 0, 0, 0};
    __builtin_amdgcn_tensor_load_to_lds(g0, g1, gz, gz, gz8, 0);
#else
    __builtin_amdgcn_tensor_load_to_lds(g0, g1, gz, gz, 0);
#endif
    __builtin_amdgcn_s_wait_tensorcnt(0);
  }
  __syncthreads();
#else
  {
    // Cooperative fallback: same padded LDS layout via regular loads/stores.
    const bf16_t* gsrc = A + m0 * K;
    for (int cchunk = tid; cchunk < 16 * ROWDW / 4; cchunk += 256) {
      int d = cchunk * 4;
      uint4 v = *(const uint4*)(gsrc + (long)d * 2);
      char* pdst = (char*)Atile + (((size_t)d << 2) + ((size_t)(d >> 6) << 4));
      *(uint4*)__builtin_assume_aligned(pdst, 16) = v;
    }
    __syncthreads();
  }
#endif

  v8f acc[NT] = {};

  for (int kk = 0; kk < K; kk += 32) {
    // A fragment (16-bit A 16x32): element e <-> K = kk+16g+e (e = 0..15).
    // Two 16B LDS reads per lane.
    int d0 = r * ROWDW + (kk + 16 * g) / 2;   // dword index of lo chunk (d0 % 4 == 0)
    int d1 = d0 + 4;                          // hi chunk
    const char* p0 = (const char*)Atile + (((unsigned)d0 << 2) + (((unsigned)d0 >> 6) << 4));
    const char* p1 = (const char*)Atile + (((unsigned)d1 << 2) + (((unsigned)d1 >> 6) << 4));
    v8bf lo = *(const v8bf*)__builtin_assume_aligned(p0, 16);
    v8bf hi = *(const v8bf*)__builtin_assume_aligned(p1, 16);
    v16bf a = __builtin_shufflevector(lo, hi, 0, 1, 2, 3, 4, 5, 6, 7,
                                      8, 9, 10, 11, 12, 13, 14, 15);
#pragma unroll
    for (int j = 0; j < NT; ++j) {
      int n0 = (wave * NT + j) * 16;
      // B fragment (32x16): lane col = r, element e <-> K = kk + e + 16g.
      const bf16_t* pb = W + (long)(n0 + r) * K + kk + 16 * g;
      v16bf b = *(const v16bf*)__builtin_assume_aligned(pb, 32);
      acc[j] = __builtin_amdgcn_wmma_f32_16x16x32_bf16(
          /*neg_a=*/false, a, /*neg_b=*/false, b,
          /*c_mod=*/(short)0, acc[j], /*reuse_a=*/false, /*reuse_b=*/false);
    }
  }

#pragma unroll
  for (int j = 0; j < NT; ++j) {
    int n0 = (wave * NT + j) * 16;
    int ccol = n0 + r;                 // C/D: lane column
    float bv = bias[ccol];
#pragma unroll
    for (int v = 0; v < 8; ++v) {
      long row = m0 + v + 8 * g;       // C/D: VGPR v <-> row v + 8g
      float val = acc[j][v] + bv;
      if (FINAL) {
        ((float*)outp)[row * DOUT + ccol] = 1.0f / (1.0f + __expf(10.0f - val));
      } else {
        val = fmaxf(val, 0.0f);        // ReLU
        ((bf16_t*)outp)[row * (5 * DOUT) + ccol] = (bf16_t)val;
      }
    }
  }
}

// ---------------------------------------------------------------------------
// Host-side orchestration. Inputs (setup_inputs order):
//  0:x [N,128] f32   1:edge_index [2,E] i32   2:edge_colour [E] i32
//  3:lin_w0 4:lin_b0 5:conv_w0   6..8: layer1   9..11: layer2
// Output: f32 [N,128].
// ---------------------------------------------------------------------------
extern "C" void kernel_launch(void* const* d_in, const int* in_sizes, int n_in,
                              void* d_out, int out_size, void* d_ws, size_t ws_size,
                              hipStream_t stream) {
  const int N = in_sizes[0] / 128;   // 50000
  const int E = in_sizes[1] / 2;     // 1600000

  const float* x    = (const float*)d_in[0];
  const int*   ei   = (const int*)d_in[1];   // src = ei[0:E], dst = ei[E:2E]
  const int*   ecol = (const int*)d_in[2];
  const float* lw0  = (const float*)d_in[3];
  const float* lb0  = (const float*)d_in[4];
  const float* cw0  = (const float*)d_in[5];
  const float* lw1  = (const float*)d_in[6];
  const float* lb1  = (const float*)d_in[7];
  const float* cw1  = (const float*)d_in[8];
  const float* lw2  = (const float*)d_in[9];
  const float* lb2  = (const float*)d_in[10];
  const float* cw2  = (const float*)d_in[11];

  char* ws = (char*)d_ws;
  size_t off = 0;
  auto alloc = [&](size_t bytes) -> void* {
    void* p = ws + off;
    off += (bytes + 255) & ~(size_t)255;
    return p;
  };

  bf16_t* A0 = (bf16_t*)alloc((size_t)N * 640 * 2);    // layer0 A: [N, 5*128]
  bf16_t* A1 = (bf16_t*)alloc((size_t)N * 1280 * 2);   // layer1 A: [N, 5*256]
  bf16_t* A2 = (bf16_t*)alloc((size_t)N * 1280 * 2);   // layer2 A: [N, 5*256]
  bf16_t* W0 = (bf16_t*)alloc((size_t)256 * 640 * 2);
  bf16_t* W1 = (bf16_t*)alloc((size_t)256 * 1280 * 2);
  bf16_t* W2 = (bf16_t*)alloc((size_t)128 * 1280 * 2);

  const int* esrc = ei;
  const int* edst = ei + E;

  auto blks = [](long total) { return (unsigned)((total + 255) / 256); };

  // ---- Layer 0 (din=128, dout=256) ----
  cvt_x_kernel<<<blks((long)N * 128), 256, 0, stream>>>(x, A0, N * 128);
  zero_agg_kernel<128><<<blks((long)N * 64), 256, 0, stream>>>(A0, N * 64);
  pack_w_kernel<128, 256><<<blks(256L * 640), 256, 0, stream>>>(lw0, cw0, W0);
  scatter_kernel<128><<<blks((long)E * 64), 256, 0, stream>>>(esrc, edst, ecol, A0, E);
  gemm_kernel<128, 256, false><<<N / 16, 256, 0, stream>>>(A0, W0, lb0, A1);

  // ---- Layer 1 (din=256, dout=256) ----
  zero_agg_kernel<256><<<blks((long)N * 128), 256, 0, stream>>>(A1, N * 128);
  pack_w_kernel<256, 256><<<blks(256L * 1280), 256, 0, stream>>>(lw1, cw1, W1);
  scatter_kernel<256><<<blks((long)E * 128), 256, 0, stream>>>(esrc, edst, ecol, A1, E);
  gemm_kernel<256, 256, false><<<N / 16, 256, 0, stream>>>(A1, W1, lb1, A2);

  // ---- Layer 2 (din=256, dout=128) ----
  zero_agg_kernel<256><<<blks((long)N * 128), 256, 0, stream>>>(A2, N * 128);
  pack_w_kernel<256, 128><<<blks(128L * 1280), 256, 0, stream>>>(lw2, cw2, W2);
  scatter_kernel<256><<<blks((long)E * 128), 256, 0, stream>>>(esrc, edst, ecol, A2, E);
  gemm_kernel<256, 128, true><<<N / 16, 256, 0, stream>>>(A2, W2, lb2, d_out);
}
